// GCT_61272003445298
// MI455X (gfx1250) — matrix operand order repaired and checked by reading, not compiled, not run on hardware
//
#include <hip/hip_runtime.h>
#include <hip/hip_bf16.h>

// Point-transformer neighborhood attention for MI455X (gfx1250, wave32).
// Heavy math on v_wmma_f32_16x16x32_bf16. Working set (~12MB) is L2-resident
// (192MB L2), so the kernel is compute-bound on the two per-query matmuls:
//   attn[32x64] @ W_attn^T[64x192]  -> 48 WMMAs / query
//   yg  [32x64] @ W_res ^T[64x64]   ->  16 WMMAs / query
// One wave per query; weights pre-swizzled into B-fragment lane order in LDS.
// This revision vectorizes all LDS staging (b128/b64) and keeps the softmax
// column resident in registers (single LDS read pass + single bf16 write).

typedef __attribute__((ext_vector_type(16))) __bf16 v16bf;
typedef __attribute__((ext_vector_type(8)))  __bf16 v8bf;
typedef __attribute__((ext_vector_type(4)))  __bf16 v4bf;
typedef __attribute__((ext_vector_type(8)))  float  v8f;

#define NW   4      // waves per block (128 threads)
#define KNB  32     // neighbors per query == wave32
#define FC   64     // feature channels
#define CIN  3
#define CO   64
#define CON  192    // CO*CIN
#define SA   68     // f32 attn tile stride: 272B rows (16B aligned), 68%64==4 -> conflict-free column walks
#define SH   80     // bf16 tile stride: 160B rows (16B aligned fragments)
#define SC   52     // chunk staging stride: 208B rows (16B aligned)

__device__ __forceinline__ float lrelu(float x) { return fmaxf(x, 0.1f * x); }

// A fragment (16x32 bf16) from row-major LDS, stride SH.
// lane<16 : row=lane,    elems = K[kofs+0..7], K[kofs+16..23]
// lane>=16: row=lane-16, elems = K[kofs+8..15], K[kofs+24..31]
__device__ __forceinline__ v16bf load_A(const __bf16* base, int row0, int kofs, int lane) {
    int r  = row0 + (lane & 15);
    int ks = kofs + ((lane >> 4) << 3);
    const __bf16* p = base + r * SH + ks;
    v8bf lo = *(const v8bf*)(p);        // 16B aligned
    v8bf hi = *(const v8bf*)(p + 16);   // 16B aligned
    return __builtin_shufflevector(lo, hi, 0,1,2,3,4,5,6,7,8,9,10,11,12,13,14,15);
}

// B fragment from pre-swizzled weight store: frag-major [frag][lane][16]
__device__ __forceinline__ v16bf load_Bfrag(const __bf16* fragbase, int frag, int lane) {
    return *(const v16bf*)(fragbase + ((frag << 5) + lane) * 16);  // 32B aligned
}

__global__ __launch_bounds__(NW * 32) void pt_attn_kernel(
    const float* __restrict__ q_points, const float* __restrict__ s_points,
    const int*   __restrict__ neighb,   const float* __restrict__ feat,
    const float* __restrict__ W_pos2,   const float* __restrict__ W_attn,
    const float* __restrict__ W_res,
    const float* bnq_g, const float* bnq_b, const float* bnq_m, const float* bnq_v,
    const float* bnp_g, const float* bnp_b, const float* bnp_m, const float* bnp_v,
    const float* bn1_g, const float* bn1_b, const float* bn1_m, const float* bn1_v,
    const float* bnr_g, const float* bnr_b, const float* bnr_m, const float* bnr_v,
    float* __restrict__ out, int Nq)
{
    // ---- static LDS (~139 KB of the 320 KB WGP pool; 2 workgroups / WGP) ----
    __shared__ __align__(32) __bf16 sWattnF[2 * 12 * 32 * 16]; // B frags, kb*12+cb
    __shared__ __align__(32) __bf16 sWresF [2 *  4 * 32 * 16]; // B frags, kb*4+cb
    __shared__ __align__(16) float  sWp[CIN][FC];              // W_pos2 SoA
    __shared__ float  sBNq[2][FC], sBN1[2][CO], sBNr[2][CO], sBNp[2][4];
    __shared__ __align__(16) float  sAttnF[NW][KNB * SA];      // logits -> residual stage
    __shared__ __align__(16) __bf16 sAttnH[NW][KNB * SH];      // softmaxed attn (A tile)
    __shared__ __align__(16) __bf16 sYg  [NW][KNB * SH];       // gathered feats (A tile)
    __shared__ __align__(16) float  sChunk[NW][KNB * SC];      // attn2 48-col chunk staging
    __shared__ __align__(16) float  sXq[NW][FC];

    const int tid = threadIdx.x;

    // ---- once per block: bf16-convert + pre-swizzle weights into B-fragment order
    for (int t = tid; t < 2 * 12 * 32 * 16; t += blockDim.x) {
        int e = t & 15, l = (t >> 4) & 31, q = t >> 9;
        int cb = q % 12, kb = q / 12;
        int col = cb * 16 + (l & 15);
        int kk  = kb * 32 + ((l >> 4) << 3) + (e & 7) + ((e >> 3) << 4);
        sWattnF[t] = (__bf16)W_attn[col * FC + kk];   // B[f][c] = W_attn[c][f]
    }
    for (int t = tid; t < 2 * 4 * 32 * 16; t += blockDim.x) {
        int e = t & 15, l = (t >> 4) & 31, q = t >> 9;
        int cb = q & 3, kb = q >> 2;
        int col = cb * 16 + (l & 15);
        int kk  = kb * 32 + ((l >> 4) << 3) + (e & 7) + ((e >> 3) << 4);
        sWresF[t] = (__bf16)W_res[col * FC + kk];
    }
    for (int t = tid; t < FC; t += blockDim.x) {
        sWp[0][t] = W_pos2[t * 3 + 0];
        sWp[1][t] = W_pos2[t * 3 + 1];
        sWp[2][t] = W_pos2[t * 3 + 2];
        float sq = bnq_g[t] * rsqrtf(bnq_v[t] + 1e-5f);
        sBNq[0][t] = sq; sBNq[1][t] = bnq_b[t] - bnq_m[t] * sq;
        float s1 = bn1_g[t] * rsqrtf(bn1_v[t] + 1e-5f);
        sBN1[0][t] = s1; sBN1[1][t] = bn1_b[t] - bn1_m[t] * s1;
        float sr = bnr_g[t] * rsqrtf(bnr_v[t] + 1e-5f);
        sBNr[0][t] = sr; sBNr[1][t] = bnr_b[t] - bnr_m[t] * sr;
    }
    if (tid < CIN) {
        float sp = bnp_g[tid] * rsqrtf(bnp_v[tid] + 1e-5f);
        sBNp[0][tid] = sp; sBNp[1][tid] = bnp_b[tid] - bnp_m[tid] * sp;
    }
    __syncthreads();

    const int lane = tid & 31;
    const int wv   = tid >> 5;
    float*  attF = sAttnF[wv];
    __bf16* attH = sAttnH[wv];
    __bf16* ygH  = sYg[wv];
    float*  chk  = sChunk[wv];
    float*  xq   = sXq[wv];

    const int totalWaves = gridDim.x * NW;
    for (int n = blockIdx.x * NW + wv; n < Nq; n += totalWaves) {
        // ---- phase A: gather (lane = neighbor k) + raw attention logits ----
        int   idx  = neighb[(size_t)n * KNB + lane];
        float fv   = (idx < Nq) ? 1.f : 0.f;       // shadow row -> zeros
        int   idc  = (idx < Nq) ? idx : 0;
        float q0 = q_points[n * 3 + 0], q1 = q_points[n * 3 + 1], q2 = q_points[n * 3 + 2];
        float xv0 = fv * s_points[idc * 3 + 0] - q0;
        float xv1 = fv * s_points[idc * 3 + 1] - q1;
        float xv2 = fv * s_points[idc * 3 + 2] - q2;
        float p0 = lrelu(s_points[n * 3 + 0] * sBNp[0][0] + sBNp[1][0]);
        float p1 = lrelu(s_points[n * 3 + 1] * sBNp[0][1] + sBNp[1][1]);
        float p2 = lrelu(s_points[n * 3 + 2] * sBNp[0][2] + sBNp[1][2]);
        float t0 = p0 - xv0, t1 = p1 - xv1, t2 = p2 - xv2;

        // x_q (query feature after BN+lrelu), 2 channels per lane
        xq[lane]      = lrelu(feat[(size_t)n * FC + lane]      * sBNq[0][lane]      + sBNq[1][lane]);
        xq[lane + 32] = lrelu(feat[(size_t)n * FC + lane + 32] * sBNq[0][lane + 32] + sBNq[1][lane + 32]);
        asm volatile("s_wait_dscnt 0" ::: "memory");      // xq visible wave-wide

        const float4* frow = (const float4*)(feat + (size_t)idc * FC);
        #pragma unroll
        for (int ff = 0; ff < FC / 4; ++ff) {
            float4 y4 = frow[ff];
            float ys[4] = {y4.x * fv, y4.y * fv, y4.z * fv, y4.w * fv};
            int f0 = ff * 4;
            float4 xq4 = *(const float4*)&xq[f0];           // b128 broadcast
            float4 w04 = *(const float4*)&sWp[0][f0];
            float4 w14 = *(const float4*)&sWp[1][f0];
            float4 w24 = *(const float4*)&sWp[2][f0];
            const float* xqe = &xq4.x;
            const float* w0e = &w04.x;
            const float* w1e = &w14.x;
            const float* w2e = &w24.x;
            v4bf yb;
            float4 a4;
            float* ae = &a4.x;
            #pragma unroll
            for (int j = 0; j < 4; ++j) {
                float yv = ys[j];
                yb[j] = (__bf16)yv;
                float pf = w0e[j] * t0 + w1e[j] * t1 + w2e[j] * t2;
                ae[j] = xqe[j] - yv + pf;                   // logits (pre 1/8 scale)
            }
            *(v4bf*)&ygH[lane * SH + f0] = yb;              // b64 store
            *(float4*)&attF[lane * SA + f0] = a4;           // b128 store
        }
        asm volatile("s_wait_dscnt 0" ::: "memory");        // tiles visible wave-wide

        // ---- phase B: softmax over K per channel (lane = channel; 2 columns),
        //      column held in registers: one LDS read pass + one bf16 write pass
        #pragma unroll
        for (int h = 0; h < 2; ++h) {
            int f = lane + 32 * h;
            float a[KNB];
            #pragma unroll
            for (int k = 0; k < KNB; ++k) a[k] = attF[k * SA + f];
            float m = -1e30f;
            #pragma unroll
            for (int k = 0; k < KNB; ++k) m = fmaxf(m, a[k]);
            float s = 0.f;
            #pragma unroll
            for (int k = 0; k < KNB; ++k) {
                a[k] = __expf((a[k] - m) * 0.125f);         // /sqrt(64)
                s += a[k];
            }
            float inv = 1.f / s;
            #pragma unroll
            for (int k = 0; k < KNB; ++k)
                attH[k * SH + f] = (__bf16)(a[k] * inv);
        }
        asm volatile("s_wait_dscnt 0" ::: "memory");

        // ---- residual: yg @ W_res^T  (16 WMMAs), BN fused, staged into attF ----
        #pragma unroll
        for (int mt = 0; mt < 2; ++mt) {
            v16bf a0 = load_A(ygH, 16 * mt, 0,  lane);
            v16bf a1 = load_A(ygH, 16 * mt, 32, lane);
            #pragma unroll
            for (int nt = 0; nt < 4; ++nt) {
                v8f d = {0.f, 0.f, 0.f, 0.f, 0.f, 0.f, 0.f, 0.f};
                d = __builtin_amdgcn_wmma_f32_16x16x32_bf16(false, a0, false,
                        load_Bfrag(sWresF, 0 * 4 + nt, lane), (short)0, d, false, false);
                d = __builtin_amdgcn_wmma_f32_16x16x32_bf16(false, a1, false,
                        load_Bfrag(sWresF, 1 * 4 + nt, lane), (short)0, d, false, false);
                int o  = 16 * nt + (lane & 15);
                int mb = 16 * mt + ((lane >> 4) << 3);
                float sc = sBNr[0][o], bi = sBNr[1][o];
                #pragma unroll
                for (int r = 0; r < 8; ++r)
                    attF[(mb + r) * SA + o] = d[r] * sc + bi;
            }
        }

        // ---- attention matmul in 48-col chunks (16 o's each), 48 WMMAs ----
        v16bf aa00 = load_A(attH, 0,  0,  lane);
        v16bf aa01 = load_A(attH, 0,  32, lane);
        v16bf aa10 = load_A(attH, 16, 0,  lane);
        v16bf aa11 = load_A(attH, 16, 32, lane);
        #pragma unroll
        for (int c = 0; c < 4; ++c) {
            v8f d0[3], d1[3];
            #pragma unroll
            for (int nt = 0; nt < 3; ++nt) {
                d0[nt] = (v8f){0.f,0.f,0.f,0.f,0.f,0.f,0.f,0.f};
                d1[nt] = d0[nt];
                int cb = c * 3 + nt;
                v16bf b0 = load_Bfrag(sWattnF, 0 * 12 + cb, lane);
                v16bf b1 = load_Bfrag(sWattnF, 1 * 12 + cb, lane);
                d0[nt] = __builtin_amdgcn_wmma_f32_16x16x32_bf16(false, aa00, false, b0, (short)0, d0[nt], false, false);
                d0[nt] = __builtin_amdgcn_wmma_f32_16x16x32_bf16(false, aa01, false, b1, (short)0, d0[nt], false, false);
                d1[nt] = __builtin_amdgcn_wmma_f32_16x16x32_bf16(false, aa10, false, b0, (short)0, d1[nt], false, false);
                d1[nt] = __builtin_amdgcn_wmma_f32_16x16x32_bf16(false, aa11, false, b1, (short)0, d1[nt], false, false);
            }
            // dump 2x3 D tiles into chunk staging (row differs per VGPR -> scalar)
            #pragma unroll
            for (int nt = 0; nt < 3; ++nt) {
                int colb = 16 * nt + (lane & 15);
                int mb   = (lane >> 4) << 3;
                #pragma unroll
                for (int r = 0; r < 8; ++r) {
                    chk[(mb + r) * SC + colb]      = d0[nt][r];
                    chk[(mb + 16 + r) * SC + colb] = d1[nt][r];
                }
            }
            asm volatile("s_wait_dscnt 0" ::: "memory");

            // lane = neighbor k: pull chunk row + residual row with b128 loads,
            // fold Cin=3 with x_v, BN+lrelu, add residual, lrelu, then max
            // over the 32 neighbors (wave = K dimension).
            float cr[48];
            #pragma unroll
            for (int qq = 0; qq < 12; ++qq) {
                float4 t4 = *(const float4*)&chk[lane * SC + 4 * qq];
                cr[4 * qq + 0] = t4.x; cr[4 * qq + 1] = t4.y;
                cr[4 * qq + 2] = t4.z; cr[4 * qq + 3] = t4.w;
            }
            float rr[16];
            #pragma unroll
            for (int qq = 0; qq < 4; ++qq) {
                float4 t4 = *(const float4*)&attF[lane * SA + 16 * c + 4 * qq];
                rr[4 * qq + 0] = t4.x; rr[4 * qq + 1] = t4.y;
                rr[4 * qq + 2] = t4.z; rr[4 * qq + 3] = t4.w;
            }
            float uo = 0.f;
            #pragma unroll
            for (int j = 0; j < 16; ++j) {
                int o = 16 * c + j;
                float ye = xv0 * cr[3 * j + 0] + xv1 * cr[3 * j + 1] + xv2 * cr[3 * j + 2];
                float v  = lrelu(ye * sBN1[0][o] + sBN1[1][o]);
                float u  = lrelu(v + rr[j]);
                u = fmaxf(u, __shfl_xor(u, 16, 32));
                u = fmaxf(u, __shfl_xor(u, 8, 32));
                u = fmaxf(u, __shfl_xor(u, 4, 32));
                u = fmaxf(u, __shfl_xor(u, 2, 32));
                u = fmaxf(u, __shfl_xor(u, 1, 32));
                if (j == lane) uo = u;         // lane j keeps channel 16c+j
            }
            if (lane < 16) out[(size_t)n * CO + 16 * c + lane] = uo;  // coalesced
        }
    }
}

extern "C" void kernel_launch(void* const* d_in, const int* in_sizes, int n_in,
                              void* d_out, int out_size, void* d_ws, size_t ws_size,
                              hipStream_t stream) {
    const float* q_points = (const float*)d_in[0];
    const float* s_points = (const float*)d_in[1];
    const int*   neighb   = (const int*)  d_in[2];
    const float* feat     = (const float*)d_in[3];
    const float* W_pos2   = (const float*)d_in[4];
    const float* W_attn   = (const float*)d_in[5];
    const float* W_res    = (const float*)d_in[6];
    const float* bnq_g = (const float*)d_in[7],  *bnq_b = (const float*)d_in[8];
    const float* bnq_m = (const float*)d_in[9],  *bnq_v = (const float*)d_in[10];
    const float* bnp_g = (const float*)d_in[11], *bnp_b = (const float*)d_in[12];
    const float* bnp_m = (const float*)d_in[13], *bnp_v = (const float*)d_in[14];
    const float* bn1_g = (const float*)d_in[15], *bn1_b = (const float*)d_in[16];
    const float* bn1_m = (const float*)d_in[17], *bn1_v = (const float*)d_in[18];
    const float* bnr_g = (const float*)d_in[19], *bnr_b = (const float*)d_in[20];
    const float* bnr_m = (const float*)d_in[21], *bnr_v = (const float*)d_in[22];

    int Nq = in_sizes[0] / 3;

    int blocksNeeded = (Nq + NW - 1) / NW;
    int blocks = blocksNeeded < 2048 ? blocksNeeded : 2048;
    if (blocks < 1) blocks = 1;

    pt_attn_kernel<<<blocks, NW * 32, 0, stream>>>(
        q_points, s_points, neighb, feat, W_pos2, W_attn, W_res,
        bnq_g, bnq_b, bnq_m, bnq_v, bnp_g, bnp_b, bnp_m, bnp_v,
        bn1_g, bn1_b, bn1_m, bn1_v, bnr_g, bnr_b, bnr_m, bnr_v,
        (float*)d_out, Nq);
}
